// PatchedEncoder_66340064854633
// MI455X (gfx1250) — matrix operand order repaired and checked by reading, not compiled, not run on hardware
//
#include <hip/hip_runtime.h>

typedef __attribute__((ext_vector_type(16))) __bf16 v16bf;
typedef __attribute__((ext_vector_type(8)))  float  v8f;

#define DD 128
#define RWSE_DIM 16
#define GG 128
#define BN_EPS 1e-5f
#define WSTRIDE 136                       // padded k-stride (ushorts)
#define WIMG_USHORTS (DD * WSTRIDE)       // 17408
#define WIMG_BYTES   (WIMG_USHORTS * 2)   // 34816 (multiple of 16)

__device__ __forceinline__ unsigned short f2bf(float f) {
  unsigned int u = __float_as_uint(f);
  u += 0x7FFFu + ((u >> 16) & 1u);        // round-to-nearest-even
  return (unsigned short)(u >> 16);
}
__device__ __forceinline__ float bf2f(unsigned short b) {
  return __uint_as_float(((unsigned int)b) << 16);
}

union FragBF { v16bf v; uint4 q[2]; unsigned short u[16]; };

// ---------------- node feature encoder -> bf16 h ----------------
__global__ void encode_kernel(const int* __restrict__ feat_id,
                              const float* __restrict__ rwse,
                              const int* __restrict__ in_deg,
                              const float* __restrict__ value_W,
                              const float* __restrict__ value_b,
                              const float* __restrict__ rwse_W,
                              const float* __restrict__ rwse_b,
                              const float* __restrict__ deg_emb,
                              unsigned short* __restrict__ hbf, int N) {
  int idx = blockIdx.x * blockDim.x + threadIdx.x;
  if (idx >= N * DD) return;
  int i = idx >> 7, d = idx & (DD - 1);
  int fid = ((unsigned)feat_id[i]) & (DD - 1);          // feat_id % 128
  float v = value_W[fid * DD + d] + value_b[d] + rwse_b[d];
  const float* rw = rwse + (size_t)i * RWSE_DIM;
  #pragma unroll
  for (int r = 0; r < RWSE_DIM; ++r) v += rw[r] * rwse_W[r * DD + d];
  int dg = in_deg[i]; dg = dg < 0 ? 0 : (dg > 1000 ? 1000 : dg);
  v += deg_emb[(size_t)dg * DD + d];
  hbf[idx] = f2bf(v);
}

// ---------------- one-time: bf16 weight images, transposed [n][k], padded ----------------
__global__ void prep_w_kernel(const float* __restrict__ W1,
                              const float* __restrict__ W2,
                              unsigned short* __restrict__ wimg) {
  int idx = blockIdx.x * blockDim.x + threadIdx.x;
  if (idx >= 4 * 2 * DD * DD) return;
  int img = idx >> 14;                  // l*2 + stage
  int t   = idx & 16383;
  int l   = img >> 1, s = img & 1;
  int k = t >> 7, n = t & 127;
  const float* W = (s ? W2 : W1) + (size_t)l * DD * DD;
  wimg[(size_t)img * WIMG_USHORTS + n * WSTRIDE + k] = f2bf(W[t]);
}

// ---------------- agg = (1+eps)*h, expand bf16 -> f32 ----------------
__global__ void init_agg_kernel(const unsigned short* __restrict__ hbf,
                                float* __restrict__ agg, int n4) {
  int i = blockIdx.x * blockDim.x + threadIdx.x;
  if (i >= n4) return;
  uint2 w = ((const uint2*)hbf)[i];
  float4 o;
  o.x = __uint_as_float(w.x << 16);
  o.y = __uint_as_float(w.x & 0xFFFF0000u);
  o.z = __uint_as_float(w.y << 16);
  o.w = __uint_as_float(w.y & 0xFFFF0000u);
  ((float4*)agg)[i] = o;
}

// ---------------- edge scatter-add: bf16 gather, f32 atomics ----------------
__global__ void edge_kernel(const int* __restrict__ src, const int* __restrict__ dst,
                            const unsigned short* __restrict__ hbf,
                            float* __restrict__ agg, int E) {
  int idx = blockIdx.x * blockDim.x + threadIdx.x;
  int e = idx >> 4;
  if (e >= E) return;
  int c = (idx & 15) << 3;                               // 8 bf16 = 16B chunk
  int s = src[e], d = dst[e];
  uint4 w = *(const uint4*)(hbf + (size_t)s * DD + c);
  float* p = agg + (size_t)d * DD + c;
  atomicAdd(p + 0, __uint_as_float(w.x << 16));
  atomicAdd(p + 1, __uint_as_float(w.x & 0xFFFF0000u));
  atomicAdd(p + 2, __uint_as_float(w.y << 16));
  atomicAdd(p + 3, __uint_as_float(w.y & 0xFFFF0000u));
  atomicAdd(p + 4, __uint_as_float(w.z << 16));
  atomicAdd(p + 5, __uint_as_float(w.z & 0xFFFF0000u));
  atomicAdd(p + 6, __uint_as_float(w.w << 16));
  atomicAdd(p + 7, __uint_as_float(w.w & 0xFFFF0000u));
}

// ---------------- fused GIN MLP: relu(BN(relu(agg@W1+b1)@W2+b2)) ----------------
__global__ __launch_bounds__(256) void gin_mlp_kernel(
    const float* __restrict__ agg,
    const unsigned short* __restrict__ W1i,   // bf16 image, [n][k] padded
    const unsigned short* __restrict__ W2i,
    const float* __restrict__ b1, const float* __restrict__ b2,
    const float* __restrict__ gmm, const float* __restrict__ bta,
    const float* __restrict__ mu_, const float* __restrict__ var_,
    unsigned short* __restrict__ houtB,       // next-layer bf16 activations
    float* __restrict__ houtF,                // final f32 output (last layer)
    int isLast, int N) {
  __shared__ __align__(16) unsigned short As[96][WSTRIDE];   // A / T tile (bf16)
  __shared__ __align__(16) unsigned short Wt[DD][WSTRIDE];   // weight [n][k] (bf16)

  const int tid  = threadIdx.x;
  const int lane = tid & 31;
  const int wave = tid >> 5;                 // wave -> 16-wide output column tile
  const int rowBase = blockIdx.x * 96;
  const unsigned ldsW = (unsigned)(unsigned long long)(void*)&Wt[0][0];

  // async DMA: W1 bf16 image -> LDS (overlaps the f32 A-tile staging below)
  for (int t = tid; t < WIMG_BYTES / 16; t += 256) {
    unsigned off = (unsigned)t * 16u;
    asm volatile("global_load_async_to_lds_b128 %0, %1, %2"
                 :: "v"(ldsW + off), "v"(off), "s"(W1i) : "memory");
  }
  // stage A tile: f32 -> bf16
  for (int t = tid; t < 96 * DD; t += 256) {
    int r = t >> 7, c = t & 127;
    int gr = rowBase + r;
    As[r][c] = (gr < N) ? f2bf(agg[(size_t)gr * DD + c]) : (unsigned short)0;
  }
  asm volatile("s_wait_asynccnt 0" ::: "memory");
  __syncthreads();

  const int ncol   = wave * 16 + (lane & 15);
  const int khalfA = (lane & 16) ? 8 : 0;    // A layout: hi lanes hold K+8
  const int khalfB = (lane & 16) ? 8 : 0;    // B layout: hi lanes hold K+8
  const int mrowL  = lane & 15;

  v8f acc[6];
  FragBF bf[4];

  // ---- stage 1: T = relu(agg @ W1 + b1) ----
  #pragma unroll
  for (int kk = 0; kk < 4; ++kk) {
    int kb = kk * 32 + khalfB;
    bf[kk].q[0] = *(const uint4*)&Wt[ncol][kb];
    bf[kk].q[1] = *(const uint4*)&Wt[ncol][kb + 16];
  }
  #pragma unroll
  for (int mt = 0; mt < 6; ++mt) {
    v8f c = {};
    #pragma unroll
    for (int kk = 0; kk < 4; ++kk) {
      int kb = kk * 32 + khalfA;
      FragBF af;
      af.q[0] = *(const uint4*)&As[mt * 16 + mrowL][kb];
      af.q[1] = *(const uint4*)&As[mt * 16 + mrowL][kb + 16];
      c = __builtin_amdgcn_wmma_f32_16x16x32_bf16(false, af.v, false, bf[kk].v,
                                                  (short)0, c, false, false);
    }
    acc[mt] = c;
  }
  __syncthreads();

  // async DMA: W2 image -> LDS (overlaps T writeback)
  for (int t = tid; t < WIMG_BYTES / 16; t += 256) {
    unsigned off = (unsigned)t * 16u;
    asm volatile("global_load_async_to_lds_b128 %0, %1, %2"
                 :: "v"(ldsW + off), "v"(off), "s"(W2i) : "memory");
  }
  {  // write T back to As as bf16
    float bb = b1[ncol];
    #pragma unroll
    for (int mt = 0; mt < 6; ++mt)
      #pragma unroll
      for (int r = 0; r < 8; ++r) {
        float v = acc[mt][r] + bb;
        v = v > 0.f ? v : 0.f;
        As[mt * 16 + r + ((lane & 16) ? 8 : 0)][ncol] = f2bf(v);
      }
  }
  asm volatile("s_wait_asynccnt 0" ::: "memory");
  __syncthreads();

  // ---- stage 2: h = relu(BN(T @ W2 + b2)) ----
  #pragma unroll
  for (int kk = 0; kk < 4; ++kk) {
    int kb = kk * 32 + khalfB;
    bf[kk].q[0] = *(const uint4*)&Wt[ncol][kb];
    bf[kk].q[1] = *(const uint4*)&Wt[ncol][kb + 16];
  }
  #pragma unroll
  for (int mt = 0; mt < 6; ++mt) {
    v8f c = {};
    #pragma unroll
    for (int kk = 0; kk < 4; ++kk) {
      int kb = kk * 32 + khalfA;
      FragBF af;
      af.q[0] = *(const uint4*)&As[mt * 16 + mrowL][kb];
      af.q[1] = *(const uint4*)&As[mt * 16 + mrowL][kb + 16];
      c = __builtin_amdgcn_wmma_f32_16x16x32_bf16(false, af.v, false, bf[kk].v,
                                                  (short)0, c, false, false);
    }
    acc[mt] = c;
  }

  {
    float bb  = b2[ncol];
    float inv = gmm[ncol] * rsqrtf(var_[ncol] + BN_EPS);
    float mu  = mu_[ncol], be = bta[ncol];
    #pragma unroll
    for (int mt = 0; mt < 6; ++mt)
      #pragma unroll
      for (int r = 0; r < 8; ++r) {
        float v = acc[mt][r] + bb;
        v = (v - mu) * inv + be;
        v = v > 0.f ? v : 0.f;
        int gr = rowBase + mt * 16 + r + ((lane & 16) ? 8 : 0);
        if (gr < N) {
          houtB[(size_t)gr * DD + ncol] = f2bf(v);
          if (isLast) houtF[(size_t)gr * DD + ncol] = v;
        }
      }
  }
}

// ---------------- pooling ----------------
__global__ void zero_kernel(float* __restrict__ pooled, float* __restrict__ cnt,
                            int np, int ng) {
  int i = blockIdx.x * blockDim.x + threadIdx.x;
  if (i < np) pooled[i] = 0.f;
  else if (i < np + ng) cnt[i - np] = 0.f;
}

__global__ void pool_kernel(const float* __restrict__ h, const int* __restrict__ batch,
                            float* __restrict__ pooled, float* __restrict__ cnt, int N) {
  int idx = blockIdx.x * blockDim.x + threadIdx.x;
  if (idx >= N * DD) return;
  int i = idx >> 7, d = idx & 127;
  int b = batch[i];
  atomicAdd(&pooled[b * DD + d], h[idx]);
  if (d == 0) atomicAdd(&cnt[b], 1.0f);
}

__global__ void finalize_kernel(const float* __restrict__ pooled,
                                const float* __restrict__ cnt,
                                float* __restrict__ out) {
  int idx = blockIdx.x * blockDim.x + threadIdx.x;
  if (idx >= GG * DD) return;
  out[idx] = pooled[idx] / fmaxf(cnt[idx >> 7], 1.0f);
}

extern "C" void kernel_launch(void* const* d_in, const int* in_sizes, int n_in,
                              void* d_out, int out_size, void* d_ws, size_t ws_size,
                              hipStream_t stream) {
  (void)n_in; (void)out_size; (void)ws_size;
  const int*   feat_id    = (const int*)d_in[0];
  const int*   edge_index = (const int*)d_in[1];
  const int*   batch_idx  = (const int*)d_in[2];
  const float* rwse       = (const float*)d_in[3];
  const int*   in_deg     = (const int*)d_in[4];
  const float* value_W    = (const float*)d_in[5];
  const float* value_b    = (const float*)d_in[6];
  const float* rwse_W     = (const float*)d_in[7];
  const float* rwse_b     = (const float*)d_in[8];
  const float* deg_emb    = (const float*)d_in[9];
  const float* mlp_W1     = (const float*)d_in[10];
  const float* mlp_b1     = (const float*)d_in[11];
  const float* mlp_W2     = (const float*)d_in[12];
  const float* mlp_b2     = (const float*)d_in[13];
  const float* bn_g       = (const float*)d_in[14];
  const float* bn_b       = (const float*)d_in[15];
  const float* bn_m       = (const float*)d_in[16];
  const float* bn_v       = (const float*)d_in[17];

  const int N = in_sizes[0];
  const int E = in_sizes[1] / 2;
  const int* src = edge_index;
  const int* dst = edge_index + E;

  // workspace layout (16B-aligned regions)
  char* ws = (char*)d_ws;
  unsigned short* hbf  = (unsigned short*)ws;                 // [N,128] bf16
  ws += ((size_t)N * DD * 2 + 15) & ~(size_t)15;
  float* agg           = (float*)ws;                          // [N,128] f32
  ws += ((size_t)N * DD * 4 + 15) & ~(size_t)15;
  unsigned short* wimg = (unsigned short*)ws;                 // 8 weight images
  ws += ((size_t)8 * WIMG_BYTES + 15) & ~(size_t)15;
  float* pooled        = (float*)ws;                          // [128,128]
  ws += (size_t)GG * DD * 4;
  float* cnt           = (float*)ws;                          // [128]

  float* out_graph = (float*)d_out;                           // [128,128]
  float* out_h     = out_graph + GG * DD;                     // [N,128]

  const int nd = N * DD;
  encode_kernel<<<(nd + 255) / 256, 256, 0, stream>>>(
      feat_id, rwse, in_deg, value_W, value_b, rwse_W, rwse_b, deg_emb, hbf, N);

  const int pw = 4 * 2 * DD * DD;
  prep_w_kernel<<<(pw + 255) / 256, 256, 0, stream>>>(mlp_W1, mlp_W2, wimg);

  for (int l = 0; l < 4; ++l) {
    const int n4 = N * (DD / 4);
    init_agg_kernel<<<(n4 + 255) / 256, 256, 0, stream>>>(hbf, agg, n4);
    const int et = E * (DD / 8);
    edge_kernel<<<(et + 255) / 256, 256, 0, stream>>>(src, dst, hbf, agg, E);
    gin_mlp_kernel<<<(N + 95) / 96, 256, 0, stream>>>(
        agg,
        wimg + (size_t)(l * 2 + 0) * WIMG_USHORTS,
        wimg + (size_t)(l * 2 + 1) * WIMG_USHORTS,
        mlp_b1 + l * DD, mlp_b2 + l * DD,
        bn_g + l * DD, bn_b + l * DD, bn_m + l * DD, bn_v + l * DD,
        hbf, out_h, (l == 3) ? 1 : 0, N);
  }

  const int zt = GG * DD + GG;
  zero_kernel<<<(zt + 255) / 256, 256, 0, stream>>>(pooled, cnt, GG * DD, GG);
  pool_kernel<<<(nd + 255) / 256, 256, 0, stream>>>(out_h, batch_idx, pooled, cnt, N);
  finalize_kernel<<<(GG * DD + 255) / 256, 256, 0, stream>>>(pooled, cnt, out_graph);
}
